// Placer_2542620639819
// MI455X (gfx1250) — compile-verified, matrix operand order checked by reading, not run on hardware
//
#include <hip/hip_runtime.h>
#include <hip/hip_bf16.h>

// ---------------------------------------------------------------------------
// Pointer-network (bi-LSTM encoder + attention decoder) for gfx1250 (CDNA5).
// All matmuls run through v_wmma_f32_16x16x32_f16 (wave32).
// GEMM: each wave computes a 16x64 strip (1 A-tile x 4 W-tiles), two clean
// K-segments, hoisted base pointers, no manual double-buffer (no spills),
// __launch_bounds__(256,1) to give the backend the full VGPR file.
// ---------------------------------------------------------------------------

#define B_   128
#define S_   256
#define DIN_ 128
#define H_   256
#define H4_  1024
#define OUT_ 20
#define TWOH_ 512

typedef __attribute__((ext_vector_type(16))) _Float16 v16h;
typedef __attribute__((ext_vector_type(8)))  _Float16 v8h;
typedef __attribute__((ext_vector_type(8)))  float    v8f;

// ---------------- small helpers -------------------------------------------

__global__ void k_cvt(const float* __restrict__ s, _Float16* __restrict__ d, int n) {
    int i = blockIdx.x * blockDim.x + threadIdx.x;
    if (i < n) d[i] = (_Float16)s[i];
}

// input_seq [B,S,D] f32 -> [S,B,D] f16
__global__ void k_xpose(const float* __restrict__ x, _Float16* __restrict__ d) {
    int i = blockIdx.x * blockDim.x + threadIdx.x;          // B*S*D
    int dd = i & (DIN_ - 1);
    int r  = i >> 7;                                        // b*S + s
    int ss = r & (S_ - 1);
    int bb = r >> 8;
    d[(ss * B_ + bb) * DIN_ + dd] = (_Float16)x[i];
}

__global__ void k_bias2(const float* __restrict__ a, const float* __restrict__ b,
                        float* __restrict__ o, int n) {
    int i = blockIdx.x * blockDim.x + threadIdx.x;
    if (i < n) o[i] = a[i] + b[i];
}

__global__ void k_fill_int(int* p, int v, int n) {
    int i = blockIdx.x * blockDim.x + threadIdx.x;
    if (i < n) p[i] = v;
}

// ---------------- WMMA GEMM ------------------------------------------------
// C[M,N] f32 = concatK(A0[M,k0], A1[M,K-k0]) @ concatK(W0[N,k0], W1[N,K-k0])^T (+bias)
// A/W are f16 row-major. One wave computes a 16x(16*NT) strip of C.
// Operand fetch matches the ISA K-striping for 16-bit A/B operands:
//   v16h elems 0..7  <- K = kbase + (lane>=16 ? 8:0) + 0..7   (one b128 load)
//   v16h elems 8..15 <- K = kbase + 16 + (lane>=16 ? 8:0) + 0..7 (one b128 load)

__device__ __forceinline__ v16h cat16(v8h lo, v8h hi) {
    return __builtin_shufflevector(lo, hi, 0, 1, 2, 3, 4, 5, 6, 7,
                                           8, 9, 10, 11, 12, 13, 14, 15);
}

template <int NT>
__global__ __launch_bounds__(256, 1)
void k_wmma_gemm(const _Float16* __restrict__ A0, int lda0,
                 const _Float16* __restrict__ A1, int lda1,
                 const _Float16* __restrict__ W0, int ldw0,
                 const _Float16* __restrict__ W1v, int ldw1,
                 int k0, int K,
                 const float* __restrict__ bias,
                 float* __restrict__ C, int M, int N) {
    const int wave     = (int)((blockIdx.x * blockDim.x + threadIdx.x) >> 5);
    const int groups_n = N / (16 * NT);
    const int total    = (M >> 4) * groups_n;
    if (wave >= total) return;                  // launches are exact; never taken

    const int tm   = wave / groups_n;
    const int tn0  = (wave % groups_n) * NT;
    const int lane = threadIdx.x & 31;
    const int half = (lane >> 4) << 3;          // 0 or 8
    const int lm   = lane & 15;

    v8f acc[NT];
#pragma unroll
    for (int i = 0; i < NT; ++i) acc[i] = (v8f){};

    auto run = [&](const _Float16* __restrict__ A, int lda,
                   const _Float16* __restrict__ W, int ldw, int Kseg) {
        if (Kseg <= 0) return;
        const _Float16* pa = A + (size_t)(tm * 16 + lm) * lda + half;
        const _Float16* pw[NT];
#pragma unroll
        for (int i = 0; i < NT; ++i)
            pw[i] = W + (size_t)((tn0 + i) * 16 + lm) * ldw + half;
        __builtin_prefetch(pa, 0, 1);

#pragma unroll 2
        for (int k = 0; k < Kseg; k += 32) {
            v16h a = cat16(*(const v8h*)(pa + k), *(const v8h*)(pa + k + 16));
#pragma unroll
            for (int i = 0; i < NT; ++i) {
                v16h w = cat16(*(const v8h*)(pw[i] + k), *(const v8h*)(pw[i] + k + 16));
                acc[i] = __builtin_amdgcn_wmma_f32_16x16x32_f16(
                    false, a, false, w, (short)0, acc[i], false, false);
            }
        }
    };

    run(A0, lda0, W0, ldw0, k0);
    run(A1, lda1, W1v, ldw1, K - k0);

    const int mb = tm * 16 + half;              // C row base for this lane half
#pragma unroll
    for (int i = 0; i < NT; ++i) {
        int n = (tn0 + i) * 16 + lm;
        float bv = bias ? bias[n] : 0.0f;
#pragma unroll
        for (int v = 0; v < 8; ++v)
            C[(size_t)(mb + v) * N + n] = acc[i][v] + bv;
    }
}

// ---------------- LSTM gate fusion ----------------------------------------

__device__ __forceinline__ float sigm(float x) { return 1.0f / (1.0f + __expf(-x)); }

// encoder fuse: z[B,4H] -> h,c update, f16 mirror, store into enc[B,S,2H]
__global__ void k_enc_fuse(const float* __restrict__ z,
                           float* __restrict__ h, float* __restrict__ c,
                           _Float16* __restrict__ h16,
                           _Float16* __restrict__ enc, int s_idx, int off) {
    int i = blockIdx.x * blockDim.x + threadIdx.x;   // B*H
    int b = i >> 8, j = i & (H_ - 1);
    const float* zr = z + (size_t)b * H4_;
    float ig = sigm(zr[j]);
    float fg = sigm(zr[H_ + j]);
    float gg = tanhf(zr[2 * H_ + j]);
    float og = sigm(zr[3 * H_ + j]);
    float cn = fg * c[i] + ig * gg;
    float hn = og * tanhf(cn);
    c[i] = cn; h[i] = hn; h16[i] = (_Float16)hn;
    enc[(size_t)(b * S_ + s_idx) * TWOH_ + off + j] = (_Float16)hn;
}

// decoder fuse: adds one-hot input projection (a column of Wih_d)
__global__ void k_dec_fuse(const float* __restrict__ z,
                           const float* __restrict__ Wd,   // [4H, OUT] f32
                           const int* __restrict__ act,
                           float* __restrict__ h, float* __restrict__ c,
                           _Float16* __restrict__ h16) {
    int i = blockIdx.x * blockDim.x + threadIdx.x;   // B*H
    int b = i >> 8, j = i & (H_ - 1);
    const float* zr = z + (size_t)b * H4_;
    float zi = zr[j], zf = zr[H_ + j], zg = zr[2 * H_ + j], zo = zr[3 * H_ + j];
    int a = act[b];
    if (a >= 0) {
        zi += Wd[(size_t)(j) * OUT_ + a];
        zf += Wd[(size_t)(H_ + j) * OUT_ + a];
        zg += Wd[(size_t)(2 * H_ + j) * OUT_ + a];
        zo += Wd[(size_t)(3 * H_ + j) * OUT_ + a];
    }
    float cn = sigm(zf) * c[i] + sigm(zi) * tanhf(zg);
    float hn = sigm(zo) * tanhf(cn);
    c[i] = cn; h[i] = hn; h16[i] = (_Float16)hn;
}

// ---------------- attention ------------------------------------------------

// u[b,s] = sum_h tanh(enc_t[b,s,h] + hW2[b,h]) * vt[h]  (one wave per (b,s))
__global__ void k_attn_scores(const float* __restrict__ enc_t,
                              const float* __restrict__ hW2,
                              const float* __restrict__ vt,
                              float* __restrict__ u) {
    int wid  = (int)((blockIdx.x * blockDim.x + threadIdx.x) >> 5);  // 0..B*S-1
    int lane = threadIdx.x & 31;
    int b = wid >> 8, s = wid & (S_ - 1);
    const float* et = enc_t + (size_t)(b * S_ + s) * H_;
    const float* hw = hW2 + (size_t)b * H_;
    float sum = 0.0f;
#pragma unroll
    for (int h = lane; h < H_; h += 32)
        sum += tanhf(et[h] + hw[h]) * vt[h];
    for (int off = 16; off; off >>= 1) sum += __shfl_down(sum, off, 32);
    if (lane == 0) u[b * S_ + s] = sum;
}

// softmax over s, then ctx[b,d] = sum_s a[s]*enc[b,s,d]   (one block per b)
__global__ void k_softmax_ctx(const float* __restrict__ u,
                              const _Float16* __restrict__ enc,
                              float* __restrict__ ctx) {
    __shared__ float a[S_];
    __shared__ float red[8];
    __shared__ float bcast;
    int b = blockIdx.x, t = threadIdx.x;
    float v = u[b * S_ + t];

    float m = v;
    for (int off = 16; off; off >>= 1) m = fmaxf(m, __shfl_xor(m, off, 32));
    if ((t & 31) == 0) red[t >> 5] = m;
    __syncthreads();
    if (t == 0) {
        float mx = red[0];
        for (int w = 1; w < 8; ++w) mx = fmaxf(mx, red[w]);
        bcast = mx;
    }
    __syncthreads();
    float e = __expf(v - bcast);
    float se = e;
    for (int off = 16; off; off >>= 1) se += __shfl_xor(se, off, 32);
    if ((t & 31) == 0) red[t >> 5] = se;
    __syncthreads();
    if (t == 0) {
        float sm = 0.0f;
        for (int w = 0; w < 8; ++w) sm += red[w];
        bcast = 1.0f / sm;
    }
    __syncthreads();
    a[t] = e * bcast;
    __syncthreads();

    const _Float16* eb = enc + (size_t)b * S_ * TWOH_;
    for (int d = t; d < TWOH_; d += blockDim.x) {
        float acc = 0.0f;
        for (int s = 0; s < S_; ++s)
            acc += a[s] * (float)eb[(size_t)s * TWOH_ + d];
        ctx[b * TWOH_ + d] = acc;
    }
}

// ---------------- logits + deterministic gumbel sampling -------------------

__device__ __forceinline__ unsigned wang(unsigned x) {
    x = (x ^ 61u) ^ (x >> 16); x *= 9u; x ^= x >> 4;
    x *= 0x27d4eb2du; x ^= x >> 15; return x;
}

// one block per b, 20 waves (640 threads): wave o computes logit o
__global__ void k_logits_sample(const float* __restrict__ ctx,
                                const float* __restrict__ h,
                                const float* __restrict__ Wp,
                                const float* __restrict__ bp,
                                const int* __restrict__ mask,
                                int step,
                                float* __restrict__ lp,
                                int* __restrict__ act,
                                float* __restrict__ out) {
    __shared__ float lg[32];
    int b = blockIdx.x;
    int wv = threadIdx.x >> 5, lane = threadIdx.x & 31;

    float sum = 0.0f;
    for (int d = lane; d < 3 * H_; d += 32) {
        float x = (d < TWOH_) ? ctx[b * TWOH_ + d] : h[b * H_ + (d - TWOH_)];
        sum += x * Wp[(size_t)wv * (3 * H_) + d];
    }
    for (int off = 16; off; off >>= 1) sum += __shfl_down(sum, off, 32);
    if (lane == 0) {
        float l = tanhf((sum + bp[wv]) * 0.1f) * 5.0f;
        if (mask[step * OUT_ + wv] == 0) l = -__builtin_inff();
        lg[wv] = l;
    }
    __syncthreads();

    if (threadIdx.x < 32) {
        float l = (lane < OUT_) ? lg[lane] : -__builtin_inff();
        float m = l;
        for (int off = 16; off; off >>= 1) m = fmaxf(m, __shfl_xor(m, off, 32));
        float e = (lane < OUT_) ? __expf(l - m) : 0.0f;
        float se = e;
        for (int off = 16; off; off >>= 1) se += __shfl_xor(se, off, 32);
        float lse = m + __logf(se);

        unsigned hsh = wang(0x9e3779b9u ^ (unsigned)(step * 131071 + b * 977 + lane));
        float uu = (float)(hsh >> 8) * (1.0f / 16777216.0f) + 1e-12f;
        float g  = -__logf(-__logf(uu));
        float val = (lane < OUT_ && l > -1e30f) ? l + g : -__builtin_inff();
        int idx = lane;
        for (int off = 16; off; off >>= 1) {
            float ov = __shfl_xor(val, off, 32);
            int   oi = __shfl_xor(idx, off, 32);
            if (ov > val) { val = ov; idx = oi; }
        }
        if (lane == 0) {
            act[b] = idx;
            float nlp = lp[b] + lg[idx] - lse;
            lp[b] = nlp;
            out[step * B_ + b]  = (float)idx;   // actions [S,B]
            out[S_ * B_ + b]    = nlp;          // log_prob [B] (final step wins)
        }
    }
}

// ---------------- host orchestration --------------------------------------

static inline size_t alignup(size_t x) { return (x + 255) & ~(size_t)255; }

extern "C" void kernel_launch(void* const* d_in, const int* in_sizes, int n_in,
                              void* d_out, int out_size, void* d_ws, size_t ws_size,
                              hipStream_t stream) {
    (void)in_sizes; (void)n_in; (void)out_size; (void)ws_size;

    const float* input_seq = (const float*)d_in[0];
    const int*   mask      = (const int*)d_in[1];
    const float* Wih_f = (const float*)d_in[2];
    const float* Whh_f = (const float*)d_in[3];
    const float* bih_f = (const float*)d_in[4];
    const float* bhh_f = (const float*)d_in[5];
    const float* Wih_b = (const float*)d_in[6];
    const float* Whh_b = (const float*)d_in[7];
    const float* bih_b = (const float*)d_in[8];
    const float* bhh_b = (const float*)d_in[9];
    const float* Wih_d = (const float*)d_in[10];
    const float* Whh_d = (const float*)d_in[11];
    const float* bih_d = (const float*)d_in[12];
    const float* bhh_d = (const float*)d_in[13];
    const float* W1    = (const float*)d_in[14];
    const float* W2    = (const float*)d_in[15];
    const float* vt    = (const float*)d_in[16];
    const float* Wp    = (const float*)d_in[17];
    const float* bp    = (const float*)d_in[18];
    float* out = (float*)d_out;

    // workspace carve-up
    char* base = (char*)d_ws;
    size_t off = 0;
    auto take = [&](size_t bytes) { void* p = base + off; off = alignup(off + bytes); return p; };

    _Float16* x16     = (_Float16*)take((size_t)S_ * B_ * DIN_ * 2);
    _Float16* wihf16  = (_Float16*)take((size_t)H4_ * DIN_ * 2);
    _Float16* whhf16  = (_Float16*)take((size_t)H4_ * H_ * 2);
    _Float16* wihb16  = (_Float16*)take((size_t)H4_ * DIN_ * 2);
    _Float16* whhb16  = (_Float16*)take((size_t)H4_ * H_ * 2);
    _Float16* whhd16  = (_Float16*)take((size_t)H4_ * H_ * 2);
    _Float16* w116    = (_Float16*)take((size_t)H_ * TWOH_ * 2);
    _Float16* w216    = (_Float16*)take((size_t)H_ * H_ * 2);
    float* biasf = (float*)take(H4_ * 4);
    float* biasb = (float*)take(H4_ * 4);
    float* biasd = (float*)take(H4_ * 4);
    float* h   = (float*)take((size_t)B_ * H_ * 4);
    float* c   = (float*)take((size_t)B_ * H_ * 4);
    _Float16* h16 = (_Float16*)take((size_t)B_ * H_ * 2);
    float* hF  = (float*)take((size_t)B_ * H_ * 4);
    float* cF  = (float*)take((size_t)B_ * H_ * 4);
    _Float16* h16F = (_Float16*)take((size_t)B_ * H_ * 2);
    float* z   = (float*)take((size_t)B_ * H4_ * 4);
    _Float16* enc16 = (_Float16*)take((size_t)B_ * S_ * TWOH_ * 2);
    float* enct = (float*)take((size_t)B_ * S_ * H_ * 4);
    float* hW2  = (float*)take((size_t)B_ * H_ * 4);
    float* u    = (float*)take((size_t)B_ * S_ * 4);
    float* ctx  = (float*)take((size_t)B_ * TWOH_ * 4);
    float* lp   = (float*)take(B_ * 4);
    int*   act  = (int*)take(B_ * 4);

    const int TB = 256;
    auto blocks = [](int n, int tb) { return (n + tb - 1) / tb; };
    // wave-exact gemm grids: waves = (M/16) * (N/64), 8 waves per block
    auto gemm_blocks = [](int M, int N) { return ((M / 16) * (N / 64)) / 8; };

    // 1) weight conversions to f16
    k_cvt<<<blocks(H4_ * DIN_, TB), TB, 0, stream>>>(Wih_f, wihf16, H4_ * DIN_);
    k_cvt<<<blocks(H4_ * H_,  TB), TB, 0, stream>>>(Whh_f, whhf16, H4_ * H_);
    k_cvt<<<blocks(H4_ * DIN_, TB), TB, 0, stream>>>(Wih_b, wihb16, H4_ * DIN_);
    k_cvt<<<blocks(H4_ * H_,  TB), TB, 0, stream>>>(Whh_b, whhb16, H4_ * H_);
    k_cvt<<<blocks(H4_ * H_,  TB), TB, 0, stream>>>(Whh_d, whhd16, H4_ * H_);
    k_cvt<<<blocks(H_ * TWOH_, TB), TB, 0, stream>>>(W1, w116, H_ * TWOH_);
    k_cvt<<<blocks(H_ * H_,   TB), TB, 0, stream>>>(W2, w216, H_ * H_);
    k_xpose<<<blocks(B_ * S_ * DIN_, TB), TB, 0, stream>>>(input_seq, x16);
    k_bias2<<<blocks(H4_, TB), TB, 0, stream>>>(bih_f, bhh_f, biasf, H4_);
    k_bias2<<<blocks(H4_, TB), TB, 0, stream>>>(bih_b, bhh_b, biasb, H4_);
    k_bias2<<<blocks(H4_, TB), TB, 0, stream>>>(bih_d, bhh_d, biasd, H4_);

    // 2) zero state
    hipMemsetAsync(h,   0, (size_t)B_ * H_ * 4, stream);
    hipMemsetAsync(c,   0, (size_t)B_ * H_ * 4, stream);
    hipMemsetAsync(h16, 0, (size_t)B_ * H_ * 2, stream);
    hipMemsetAsync(lp,  0, B_ * 4, stream);
    k_fill_int<<<1, B_, 0, stream>>>(act, -1, B_);

    const int fuse_blocks = (B_ * H_) / TB;                      // 128

    // 3) forward encoder LSTM: z = [x|h] @ [Wih|Whh]^T + bias
    for (int s = 0; s < S_; ++s) {
        k_wmma_gemm<4><<<gemm_blocks(B_, H4_), TB, 0, stream>>>(
            x16 + (size_t)s * B_ * DIN_, DIN_, h16, H_,
            wihf16, DIN_, whhf16, H_,
            DIN_, DIN_ + H_, biasf, z, B_, H4_);
        k_enc_fuse<<<fuse_blocks, TB, 0, stream>>>(z, h, c, h16, enc16, s, 0);
    }

    // snapshot forward finals for the decoder
    hipMemcpyAsync(hF,   h,   (size_t)B_ * H_ * 4, hipMemcpyDeviceToDevice, stream);
    hipMemcpyAsync(cF,   c,   (size_t)B_ * H_ * 4, hipMemcpyDeviceToDevice, stream);
    hipMemcpyAsync(h16F, h16, (size_t)B_ * H_ * 2, hipMemcpyDeviceToDevice, stream);

    // 4) backward encoder LSTM
    hipMemsetAsync(h,   0, (size_t)B_ * H_ * 4, stream);
    hipMemsetAsync(c,   0, (size_t)B_ * H_ * 4, stream);
    hipMemsetAsync(h16, 0, (size_t)B_ * H_ * 2, stream);
    for (int t = 0; t < S_; ++t) {
        int s = S_ - 1 - t;
        k_wmma_gemm<4><<<gemm_blocks(B_, H4_), TB, 0, stream>>>(
            x16 + (size_t)s * B_ * DIN_, DIN_, h16, H_,
            wihb16, DIN_, whhb16, H_,
            DIN_, DIN_ + H_, biasb, z, B_, H4_);
        k_enc_fuse<<<fuse_blocks, TB, 0, stream>>>(z, h, c, h16, enc16, s, H_);
    }

    // 5) enc_t = enc @ W1^T   [B*S, 512] x [256, 512]^T
    k_wmma_gemm<4><<<gemm_blocks(B_ * S_, H_), TB, 0, stream>>>(
        enc16, TWOH_, enc16, TWOH_,
        w116, TWOH_, w116, TWOH_,
        TWOH_, TWOH_, nullptr, enct, B_ * S_, H_);

    // 6) decoder
    hipMemcpyAsync(h,   hF,   (size_t)B_ * H_ * 4, hipMemcpyDeviceToDevice, stream);
    hipMemcpyAsync(c,   cF,   (size_t)B_ * H_ * 4, hipMemcpyDeviceToDevice, stream);
    hipMemcpyAsync(h16, h16F, (size_t)B_ * H_ * 2, hipMemcpyDeviceToDevice, stream);

    const int score_blocks = (B_ * S_) / 8;      // one wave per (b,s), 8/block
    for (int t = 0; t < S_; ++t) {
        // recurrent projection (one-hot input handled as a column add in fuse)
        k_wmma_gemm<4><<<gemm_blocks(B_, H4_), TB, 0, stream>>>(
            h16, H_, h16, H_,
            whhd16, H_, whhd16, H_,
            H_, H_, biasd, z, B_, H4_);
        k_dec_fuse<<<fuse_blocks, TB, 0, stream>>>(z, Wih_d, act, h, c, h16);

        // hW2 = h @ W2^T
        k_wmma_gemm<4><<<gemm_blocks(B_, H_), TB, 0, stream>>>(
            h16, H_, h16, H_,
            w216, H_, w216, H_,
            H_, H_, nullptr, hW2, B_, H_);

        k_attn_scores<<<score_blocks, TB, 0, stream>>>(enct, hW2, vt, u);
        k_softmax_ctx<<<B_, TB, 0, stream>>>(u, enc16, ctx);
        k_logits_sample<<<B_, 32 * OUT_, 0, stream>>>(ctx, h, Wp, bp, mask, t, lp, act, out);
    }
}
